// nnmodel_15951508538104
// MI455X (gfx1250) — matrix operand order repaired and checked by reading, not compiled, not run on hardware
//
#include <hip/hip_runtime.h>

// Problem constants (fixed by the reference)
#define NDIM   4096
#define BATCH  2048
#define HID_F  5
#define KDIM   (NDIM * HID_F)      // 20480

typedef __attribute__((ext_vector_type(16))) __bf16 v16bf;
typedef __attribute__((ext_vector_type(8)))  __bf16 v8bf;
typedef __attribute__((ext_vector_type(8)))  float  v8f;

// ---------------------------------------------------------------------------
// Stage 1: z[b, n*5+f] = relu( (sum_{d=-5..4} x[b,(n+d)%N]) * W_rel[f] + b_rel[f] )
// stored as bf16, row-major (B x 20480), K contiguous.
// ---------------------------------------------------------------------------
__global__ __launch_bounds__(256)
void build_z_kernel(const float* __restrict__ x,
                    const float* __restrict__ W_rel,
                    const float* __restrict__ b_rel,
                    __bf16* __restrict__ Z)
{
    __shared__ float xs[NDIM];
    const int b   = blockIdx.x;
    const int tid = threadIdx.x;

    const float4* xr4 = (const float4*)(x + (size_t)b * NDIM);
    float4* xs4 = (float4*)xs;
#pragma unroll
    for (int i = 0; i < NDIM / 4 / 256; ++i)
        xs4[tid + 256 * i] = xr4[tid + 256 * i];
    __syncthreads();

    float wr[HID_F], br[HID_F];
#pragma unroll
    for (int f = 0; f < HID_F; ++f) { wr[f] = W_rel[f]; br[f] = b_rel[f]; }

    __bf16* zrow = Z + (size_t)b * KDIM;
    const int n0 = tid * (NDIM / 256);

    float agg = 0.f;
#pragma unroll
    for (int d = -5; d < 5; ++d) agg += xs[(n0 + d) & (NDIM - 1)];

#pragma unroll
    for (int i = 0; i < NDIM / 256; ++i) {
        const int n = n0 + i;
#pragma unroll
        for (int f = 0; f < HID_F; ++f) {
            float zv = fmaf(agg, wr[f], br[f]);
            zrow[n * HID_F + f] = (__bf16)(zv > 0.f ? zv : 0.f);
        }
        agg += xs[(n + 5) & (NDIM - 1)] - xs[(n - 5) & (NDIM - 1)];
    }
}

// ---------------------------------------------------------------------------
// One-shot fp32 -> bf16 conversion of W_final (runs once per launch; pays
// 335MB+168MB once to save ~1.3 GB of HBM traffic across the 8 GEMM M-passes).
// ---------------------------------------------------------------------------
__global__ __launch_bounds__(256)
void convert_w_kernel(const float* __restrict__ Wf, __bf16* __restrict__ Wb)
{
    const size_t i = ((size_t)blockIdx.x * 256 + threadIdx.x) * 8;
    float4 a = *(const float4*)(Wf + i);
    float4 b = *(const float4*)(Wf + i + 4);
    v8bf o;
    o[0] = (__bf16)a.x; o[1] = (__bf16)a.y; o[2] = (__bf16)a.z; o[3] = (__bf16)a.w;
    o[4] = (__bf16)b.x; o[5] = (__bf16)b.y; o[6] = (__bf16)b.z; o[7] = (__bf16)b.w;
    *(v8bf*)(Wb + i) = o;
}

// ---------------------------------------------------------------------------
// GEMM tiling: Y (2048 x 4096) = Z (2048 x 20480 bf16) * W^T
// Block tile 256x128, K-tile 64; 512 threads = 16 waves (4M x 4N),
// wave tile 64x32 = 4x2 WMMA f32 accumulators.
// ---------------------------------------------------------------------------
#define BM       256
#define BN       128
#define BKK      64
#define LDK      (BKK + 8)          // 144B row stride: 16B aligned, bank-spread
#define NTHREADS 512
#define A_CH     ((BM * BKK / 8) / NTHREADS)   // 4 async b128 per lane
#define B_CH     ((BN * BKK / 8) / NTHREADS)   // 2 async b128 per lane

// CDNA5 async global->LDS copy (16B per lane), tracked by ASYNCcnt.
__device__ __forceinline__ void async_cp16(unsigned lds_addr, const __bf16* g)
{
    asm volatile("global_load_async_to_lds_b128 %0, %1, off"
                 :: "v"(lds_addr), "v"(g) : "memory");
}

// ---- Preferred path: bf16 W, async-to-LDS staging, double-buffered LDS ----
__global__ __launch_bounds__(NTHREADS, 1)
void gemm_bf16_async_kernel(const __bf16* __restrict__ Z,
                            const __bf16* __restrict__ Wb,
                            float* __restrict__ Y)
{
    __shared__ __bf16 As[2][BM][LDK];   // 2 x 36 KiB
    __shared__ __bf16 Bs[2][BN][LDK];   // 2 x 18 KiB

    const int tid  = threadIdx.x;
    const int lane = tid & 31;
    const int wave = tid >> 5;
    const int wm   = wave >> 2;         // 0..3 : 64-row slab
    const int wn   = wave & 3;          // 0..3 : 32-col slab
    const int l16  = lane & 15;
    const int hi   = lane >> 4;

    const int bm = blockIdx.y * BM;
    const int bn = blockIdx.x * BN;

    v8f acc[4][2] = {};

    auto issue = [&](int buf, int k0) {
#pragma unroll
        for (int i = 0; i < A_CH; ++i) {
            int seg = tid + NTHREADS * i;           // 8 x 16B segs per row
            int r = seg >> 3, c = (seg & 7) * 8;
            async_cp16((unsigned)(size_t)&As[buf][r][c],
                       Z + (size_t)(bm + r) * KDIM + k0 + c);
        }
#pragma unroll
        for (int i = 0; i < B_CH; ++i) {
            int seg = tid + NTHREADS * i;
            int r = seg >> 3, c = (seg & 7) * 8;
            async_cp16((unsigned)(size_t)&Bs[buf][r][c],
                       Wb + (size_t)(bn + r) * KDIM + k0 + c);
        }
    };

    const int ntile = KDIM / BKK;
    issue(0, 0);

    for (int t = 0; t < ntile; ++t) {
        const int buf = t & 1;
        if (t + 1 < ntile) {
            issue(buf ^ 1, (t + 1) * BKK);
            // group t done when <= (A_CH+B_CH)=6 of group t+1 remain in flight
            asm volatile("s_wait_asynccnt 0x6" ::: "memory");
        } else {
            asm volatile("s_wait_asynccnt 0x0" ::: "memory");
        }
        __syncthreads();

#pragma unroll
        for (int ks = 0; ks < BKK / 32; ++ks) {
            const int kk = ks * 32;
            const int ka = kk + hi * 8;
            const int kb = kk + hi * 8;
            v16bf a[4], bb[2];
#pragma unroll
            for (int mt = 0; mt < 4; ++mt) {
                const int r = wm * 64 + mt * 16 + l16;
                v8bf lo = *(const v8bf*)(&As[buf][r][ka]);
                v8bf h  = *(const v8bf*)(&As[buf][r][ka + 16]);
                a[mt] = __builtin_shufflevector(lo, h,
                        0,1,2,3,4,5,6,7,8,9,10,11,12,13,14,15);
            }
#pragma unroll
            for (int nt = 0; nt < 2; ++nt) {
                const int n = wn * 32 + nt * 16 + l16;
                v8bf lo = *(const v8bf*)(&Bs[buf][n][kb]);
                v8bf h  = *(const v8bf*)(&Bs[buf][n][kb + 16]);
                bb[nt] = __builtin_shufflevector(lo, h,
                         0,1,2,3,4,5,6,7,8,9,10,11,12,13,14,15);
            }
#pragma unroll
            for (int mt = 0; mt < 4; ++mt)
#pragma unroll
                for (int nt = 0; nt < 2; ++nt)
                    acc[mt][nt] = __builtin_amdgcn_wmma_f32_16x16x32_bf16(
                        false, a[mt], false, bb[nt],
                        (short)0, acc[mt][nt], false, false);
        }
        __syncthreads();   // all waves done reading buf before it is re-filled
    }

#pragma unroll
    for (int mt = 0; mt < 4; ++mt) {
#pragma unroll
        for (int nt = 0; nt < 2; ++nt) {
            const int col  = bn + wn * 32 + nt * 16 + l16;
            const int row0 = bm + wm * 64 + mt * 16 + hi * 8;
#pragma unroll
            for (int r = 0; r < 8; ++r)
                Y[(size_t)(row0 + r) * NDIM + col] = acc[mt][nt][r];
        }
    }
}

// ---- Fallback path (small workspace): fp32 W converted while staging ----
__global__ __launch_bounds__(NTHREADS, 1)
void gemm_bf16_sync_kernel(const __bf16* __restrict__ Z,
                           const float* __restrict__ Wf,
                           float* __restrict__ Y)
{
    __shared__ __bf16 As[BM][LDK];
    __shared__ __bf16 Bs[BN][LDK];

    const int tid  = threadIdx.x;
    const int lane = tid & 31;
    const int wave = tid >> 5;
    const int wm   = wave >> 2;
    const int wn   = wave & 3;
    const int l16  = lane & 15;
    const int hi   = lane >> 4;

    const int bm = blockIdx.y * BM;
    const int bn = blockIdx.x * BN;

    v8f acc[4][2] = {};

    for (int k0 = 0; k0 < KDIM; k0 += BKK) {
#pragma unroll
        for (int i = 0; i < A_CH; ++i) {
            int seg = tid + NTHREADS * i;
            int r = seg >> 3, c = (seg & 7) * 8;
            *(v8bf*)(&As[r][c]) =
                *(const v8bf*)(Z + (size_t)(bm + r) * KDIM + k0 + c);
        }
#pragma unroll
        for (int i = 0; i < (BN * BKK / 4) / NTHREADS; ++i) {
            int seg = tid + NTHREADS * i;
            int r = seg >> 4, c = (seg & 15) * 4;
            float4 v = *(const float4*)(Wf + (size_t)(bn + r) * KDIM + k0 + c);
            __bf16* d = &Bs[r][c];
            d[0] = (__bf16)v.x; d[1] = (__bf16)v.y;
            d[2] = (__bf16)v.z; d[3] = (__bf16)v.w;
        }
        if (k0 + BKK < KDIM) {
            int ra = tid >> 3, ca = (tid & 7) * 8;
            __builtin_prefetch(Z + (size_t)(bm + ra) * KDIM + k0 + BKK + ca, 0, 1);
            int rb = tid >> 4, cb = (tid & 15) * 4;
            __builtin_prefetch(Wf + (size_t)(bn + rb) * KDIM + k0 + BKK + cb, 0, 1);
        }
        __syncthreads();

#pragma unroll
        for (int ks = 0; ks < BKK / 32; ++ks) {
            const int kk = ks * 32;
            const int ka = kk + hi * 8;
            const int kb = kk + hi * 8;
            v16bf a[4], bb[2];
#pragma unroll
            for (int mt = 0; mt < 4; ++mt) {
                const int r = wm * 64 + mt * 16 + l16;
                v8bf lo = *(const v8bf*)(&As[r][ka]);
                v8bf h  = *(const v8bf*)(&As[r][ka + 16]);
                a[mt] = __builtin_shufflevector(lo, h,
                        0,1,2,3,4,5,6,7,8,9,10,11,12,13,14,15);
            }
#pragma unroll
            for (int nt = 0; nt < 2; ++nt) {
                const int n = wn * 32 + nt * 16 + l16;
                v8bf lo = *(const v8bf*)(&Bs[n][kb]);
                v8bf h  = *(const v8bf*)(&Bs[n][kb + 16]);
                bb[nt] = __builtin_shufflevector(lo, h,
                         0,1,2,3,4,5,6,7,8,9,10,11,12,13,14,15);
            }
#pragma unroll
            for (int mt = 0; mt < 4; ++mt)
#pragma unroll
                for (int nt = 0; nt < 2; ++nt)
                    acc[mt][nt] = __builtin_amdgcn_wmma_f32_16x16x32_bf16(
                        false, a[mt], false, bb[nt],
                        (short)0, acc[mt][nt], false, false);
        }
        __syncthreads();
    }

#pragma unroll
    for (int mt = 0; mt < 4; ++mt) {
#pragma unroll
        for (int nt = 0; nt < 2; ++nt) {
            const int col  = bn + wn * 32 + nt * 16 + l16;
            const int row0 = bm + wm * 64 + mt * 16 + hi * 8;
#pragma unroll
            for (int r = 0; r < 8; ++r)
                Y[(size_t)(row0 + r) * NDIM + col] = acc[mt][nt][r];
        }
    }
}

// ---------------------------------------------------------------------------
extern "C" void kernel_launch(void* const* d_in, const int* in_sizes, int n_in,
                              void* d_out, int out_size, void* d_ws, size_t ws_size,
                              hipStream_t stream)
{
    (void)in_sizes; (void)n_in; (void)out_size;

    const float* x     = (const float*)d_in[0];
    const float* W_rel = (const float*)d_in[1];
    const float* b_rel = (const float*)d_in[2];
    // d_in[3] = W_root: multiplied by an all-zero tensor in the reference -> unused.
    const float* W_fin = (const float*)d_in[4];
    // d_in[5]/d_in[6] = edges: fixed circular stride-5 window, fused analytically.

    float*  Y = (float*)d_out;

    const size_t zBytes = (size_t)BATCH * KDIM * 2;   // 80 MiB
    const size_t wBytes = (size_t)NDIM  * KDIM * 2;   // 160 MiB

    __bf16* Z = (__bf16*)d_ws;
    build_z_kernel<<<BATCH, 256, 0, stream>>>(x, W_rel, b_rel, Z);

    dim3 grid(NDIM / BN, BATCH / BM);   // 32 x 8

    if (ws_size >= zBytes + wBytes) {
        __bf16* Wb = (__bf16*)((char*)d_ws + zBytes);
        convert_w_kernel<<<(int)(((size_t)NDIM * KDIM / 8) / 256), 256, 0, stream>>>(W_fin, Wb);
        gemm_bf16_async_kernel<<<grid, NTHREADS, 0, stream>>>(Z, Wb, Y);
    } else {
        gemm_bf16_sync_kernel<<<grid, NTHREADS, 0, stream>>>(Z, W_fin, Y);
    }
}
